// clifford_SE3Hyena_34308198761112
// MI455X (gfx1250) — matrix operand order repaired and checked by reading, not compiled, not run on hardware
//
#include <hip/hip_runtime.h>
#include <math.h>

typedef _Float16 h16 __attribute__((ext_vector_type(16)));
typedef _Float16 h2  __attribute__((ext_vector_type(2)));
typedef float    f8  __attribute__((ext_vector_type(8)));

#define B_  4
#define N_  4096
#define D_  256
#define BN_ (B_*N_)

static __device__ __forceinline__ f8 wmma16(h16 a, h16 b, f8 c) {
  // D = A(16x32,f16) * B(32x16,f16) + C(16x16,f32)
  return __builtin_amdgcn_wmma_f32_16x16x32_f16(false, a, false, b, (short)0, c, false, false);
}

// ---------------------------------------------------------------------------
// Kernel 0: transpose + convert the three invariant weights to f16.
// Wt[n*D + k] = (f16) W[k*D + n]  -> B-fragment loads become contiguous pairs.
// ---------------------------------------------------------------------------
__global__ __launch_bounds__(256) void wcvt_kernel(
    const float* __restrict__ Wqi, const float* __restrict__ Wki, const float* __restrict__ Wvi,
    _Float16* __restrict__ Tqi, _Float16* __restrict__ Tki, _Float16* __restrict__ Tvi)
{
  const int i = blockIdx.x * 256 + threadIdx.x;   // over D*D
  const int k = i >> 8;
  const int n = i & 255;
  const size_t o = (size_t)n * D_ + k;
  Tqi[o] = (_Float16)Wqi[i];
  Tki[o] = (_Float16)Wki[i];
  Tvi[o] = (_Float16)Wvi[i];
}

// ---------------------------------------------------------------------------
// GEMM core: one wave computes acc16[nc] = A(16x256) * Wt-column-tile nc.
// All 8 B-fragments are fetched up front (one clause of b128 loads) so the
// 8-WMMA chain overlaps the loads instead of waiting loadcnt==0 each step.
// ---------------------------------------------------------------------------
static __device__ __forceinline__ f8 gemm_col(
    const h16 (&af)[8], const _Float16* __restrict__ wr, int bkoff)
{
  h16 bf[8];
  #pragma unroll
  for (int kc = 0; kc < 8; ++kc) {
    #pragma unroll
    for (int r = 0; r < 8; ++r) {
      const h2 p = *(const h2*)(wr + kc*32 + (bkoff >> 1) + 2*r + (r >= 4 ? 8 : 0));
      bf[kc][2*r] = p[0]; bf[kc][2*r+1] = p[1];
    }
  }
  f8 acc;
  #pragma unroll
  for (int r = 0; r < 8; ++r) acc[r] = 0.f;
  #pragma unroll
  for (int kc = 0; kc < 8; ++kc) acc = wmma16(af[kc], bf[kc], acc);
  return acc;
}

static __device__ __forceinline__ void gemm_tile_h(
    const h16 (&af)[8], const _Float16* __restrict__ Wt,
    int lane, int qbase, _Float16* __restrict__ out)
{
  const int m16   = lane & 15;
  const int half  = lane >> 4;
  const int bkoff = half * 16;
  for (int nc = 0; nc < 16; ++nc) {
    const int n = nc*16 + m16;
    const f8 acc = gemm_col(af, Wt + (size_t)n * D_, bkoff);
    #pragma unroll
    for (int r = 0; r < 8; ++r)
      out[(size_t)(qbase + r + half*8) * D_ + n] = (_Float16)acc[r];
  }
}

static __device__ __forceinline__ void gemm_tile_f(
    const h16 (&af)[8], const _Float16* __restrict__ Wt,
    int lane, int qbase, float* __restrict__ out)
{
  const int m16   = lane & 15;
  const int half  = lane >> 4;
  const int bkoff = half * 16;
  for (int nc = 0; nc < 16; ++nc) {
    const int n = nc*16 + m16;
    const f8 acc = gemm_col(af, Wt + (size_t)n * D_, bkoff);
    #pragma unroll
    for (int r = 0; r < 8; ++r)
      out[(size_t)(qbase + r + half*8) * D_ + n] = acc[r];
  }
}

// ---------------------------------------------------------------------------
// Kernel 1: scalar heads + q/k/v equivariant gating + 3 invariant GEMMs.
// 256 threads = 8 waves, 128 rows per block.
// ---------------------------------------------------------------------------
__global__ __launch_bounds__(256) void proj_kernel(
    const float* __restrict__ x,   const float* __restrict__ f,
    const float* __restrict__ Wqe, const float* __restrict__ Wke, const float* __restrict__ Wve,
    const _Float16* __restrict__ Tqi, const _Float16* __restrict__ Tki, const _Float16* __restrict__ Tvi,
    float* __restrict__ q_eqv, float* __restrict__ k_eqv, float* __restrict__ v_eqv,
    _Float16* __restrict__ qi16, _Float16* __restrict__ ki16, float* __restrict__ v_inv)
{
  const int tid = threadIdx.x;
  const int rb  = blockIdx.x * 128;

  if (tid < 128) {                       // scalar heads: one row per thread
    const int row = rb + tid;
    const float* fr = f + (size_t)row * D_;
    float sq = 0.f, sk = 0.f, sv = 0.f;
    for (int d = 0; d < D_; ++d) {
      const float fv = fr[d];
      sq = fmaf(fv, Wqe[d], sq);
      sk = fmaf(fv, Wke[d], sk);
      sv = fmaf(fv, Wve[d], sv);
    }
    #pragma unroll
    for (int c = 0; c < 3; ++c) {
      const float xv = x[(size_t)row*3 + c];
      q_eqv[(size_t)row*3 + c] = xv * sq;
      k_eqv[(size_t)row*3 + c] = xv * sk;
      v_eqv[(size_t)row*3 + c] = xv * sv;
    }
  }

  const int lane  = tid & 31;
  const int wave  = tid >> 5;
  const int m16   = lane & 15;
  const int half  = lane >> 4;
  const int qbase = rb + wave*16;
  const int akoff = half * 8;

  h16 af[8];                             // A fragments of f (f32 -> f16)
  {
    const float* frow = f + (size_t)(qbase + m16) * D_;
    #pragma unroll
    for (int kc = 0; kc < 8; ++kc) {
      #pragma unroll
      for (int r = 0; r < 4; ++r) {
        const int d = kc*32 + akoff + 2*r;
        af[kc][2*r]     = (_Float16)frow[d];
        af[kc][2*r+1]   = (_Float16)frow[d+1];
        af[kc][8+2*r]   = (_Float16)frow[d+16];
        af[kc][8+2*r+1] = (_Float16)frow[d+17];
      }
    }
  }
  gemm_tile_h(af, Tqi, lane, qbase, qi16);
  gemm_tile_h(af, Tki, lane, qbase, ki16);
  gemm_tile_f(af, Tvi, lane, qbase, v_inv);
}

// ---------------------------------------------------------------------------
// Kernel 2: circular convolution along N (== irfft(rfft(q)*rfft(k))).
// One (b,c,segment) per block; full channels staged in LDS (2 x 16 KB).
// ---------------------------------------------------------------------------
__global__ __launch_bounds__(256) void conv_kernel(
    const float* __restrict__ q_eqv, const float* __restrict__ k_eqv,
    float* __restrict__ u_eqv)
{
  __shared__ float qs[N_];
  __shared__ float ks[N_];
  const int bc  = blockIdx.x >> 3;   // 0..11 -> (b,c)
  const int seg = blockIdx.x & 7;    // output segment of 512
  const int b   = bc / 3;
  const int c   = bc % 3;
  const float* qp = q_eqv + (size_t)b * N_ * 3 + c;
  const float* kp = k_eqv + (size_t)b * N_ * 3 + c;
  for (int i = threadIdx.x; i < N_; i += 256) {
    qs[i] = qp[(size_t)i * 3];
    ks[i] = kp[(size_t)i * 3];
  }
  __syncthreads();
  const int n0 = seg*512 + threadIdx.x;
  const int n1 = n0 + 256;
  float acc0 = 0.f, acc1 = 0.f;
  for (int m = 0; m < N_; ++m) {
    const float qv = qs[m];
    acc0 = fmaf(qv, ks[(n0 - m) & (N_-1)], acc0);
    acc1 = fmaf(qv, ks[(n1 - m) & (N_-1)], acc1);
  }
  u_eqv[((size_t)b*N_ + n0)*3 + c] = acc0;
  u_eqv[((size_t)b*N_ + n1)*3 + c] = acc1;
}

// ---------------------------------------------------------------------------
// Kernel 3: flash attention, u_inv = softmax(Q Kᵀ / 16) K.
// 128 threads = 4 waves; wave owns 16 query rows x full D=256.
// Double-buffered 32-key K tiles in LDS + L2 prefetch of tile t+2.
// ---------------------------------------------------------------------------
#define KT  32
#define LDK 264   // padded LDS row stride (halfs) for the K tile
#define PJ  34    // padded row stride for the P scratch

__global__ __launch_bounds__(128) void attn_kernel(
    const _Float16* __restrict__ qi, const _Float16* __restrict__ ki,
    float* __restrict__ u_inv)
{
  __shared__ __align__(16) _Float16 kt[2][KT * LDK];
  __shared__ _Float16 ps[4][16 * PJ];

  const int tid   = threadIdx.x;
  const int lane  = tid & 31;
  const int wave  = tid >> 5;
  const int m16   = lane & 15;
  const int half  = lane >> 4;
  const int b     = blockIdx.x >> 6;                      // 64 q-tiles per batch
  const int qb0   = ((blockIdx.x & 63) << 6) + wave*16;   // this wave's 16 rows
  const int akoff = half * 8;
  const int bkoff = half * 16;

  const _Float16* qbp = qi + (size_t)b * N_ * D_;
  const _Float16* kbp = ki + (size_t)b * N_ * D_;

  // Q fragments, softmax scale 1/sqrt(256)=1/16 folded in exactly (pow2)
  h16 af[8];
  {
    const _Float16* qr = qbp + (size_t)(qb0 + m16) * D_;
    const _Float16 sc = (_Float16)0.0625f;
    #pragma unroll
    for (int kc = 0; kc < 8; ++kc) {
      #pragma unroll
      for (int r = 0; r < 4; ++r) {
        const int d = kc*32 + akoff + 2*r;
        af[kc][2*r]     = qr[d]    * sc;
        af[kc][2*r+1]   = qr[d+1]  * sc;
        af[kc][8+2*r]   = qr[d+16] * sc;
        af[kc][8+2*r+1] = qr[d+17] * sc;
      }
    }
  }

  f8 Oa[16];
  #pragma unroll
  for (int i = 0; i < 16; ++i) {
    #pragma unroll
    for (int r = 0; r < 8; ++r) Oa[i][r] = 0.f;
  }
  float mrow[8], lrow[8];
  #pragma unroll
  for (int r = 0; r < 8; ++r) { mrow[r] = -1e30f; lrow[r] = 0.f; }

  _Float16* pw = &ps[wave][0];

  auto stage = [&](int buf, int t) {
    const _Float16* src = kbp + (size_t)t * KT * D_;
    _Float16* dst = &kt[buf][0];
    for (int i = tid; i < (KT*D_)/8; i += 128) {          // 16B chunks
      const int e = i * 8;
      *(uint4*)(dst + (e >> 8)*LDK + (e & 255)) = *(const uint4*)(src + e);
    }
  };

  const int T = N_/KT;
  stage(0, 0);
  for (int t = 0; t < T; ++t) {
    __syncthreads();   // buf[t&1] fully staged; buf[(t+1)&1] free for reuse
    if (t + 2 < T) {   // warm L2 for the tile after next (global_prefetch_b8)
      __builtin_prefetch(kbp + (size_t)(t + 2) * KT * D_ + (size_t)tid * 64, 0, 1);
    }
    if (t + 1 < T) stage((t + 1) & 1, t + 1);
    const _Float16* kcur = &kt[t & 1][0];

    // scores S = Q x K_tileᵀ  (two 16x16 sub-tiles over 32 keys)
    f8 S0, S1;
    #pragma unroll
    for (int r = 0; r < 8; ++r) { S0[r] = 0.f; S1[r] = 0.f; }
    #pragma unroll
    for (int kc = 0; kc < 8; ++kc) {
      h16 b0, b1;
      const _Float16* k0 = kcur + (size_t)m16        * LDK + kc*32 + bkoff;
      const _Float16* k1 = kcur + (size_t)(16 + m16) * LDK + kc*32 + bkoff;
      #pragma unroll
      for (int r = 0; r < 8; ++r) {
        const h2 p0 = *(const h2*)(k0 - (bkoff >> 1) + 2*r + (r >= 4 ? 8 : 0));
        b0[2*r] = p0[0]; b0[2*r+1] = p0[1];
        const h2 p1 = *(const h2*)(k1 - (bkoff >> 1) + 2*r + (r >= 4 ? 8 : 0));
        b1[2*r] = p1[0]; b1[2*r+1] = p1[1];
      }
      S0 = wmma16(af[kc], b0, S0);
      S1 = wmma16(af[kc], b1, S1);
    }

    // online softmax: row stats replicated per 16-lane half
    float alpha[8];
    #pragma unroll
    for (int r = 0; r < 8; ++r) {
      float v = fmaxf(S0[r], S1[r]);
      v = fmaxf(v, __shfl_xor(v, 1, 32));
      v = fmaxf(v, __shfl_xor(v, 2, 32));
      v = fmaxf(v, __shfl_xor(v, 4, 32));
      v = fmaxf(v, __shfl_xor(v, 8, 32));
      const float nm = fmaxf(mrow[r], v);
      alpha[r] = __expf(mrow[r] - nm);
      const float p0 = __expf(S0[r] - nm);
      const float p1 = __expf(S1[r] - nm);
      const int mr = r + akoff;                // this lane's row = r + half*8
      pw[mr*PJ + m16]      = (_Float16)p0;     // P transpose via LDS scratch
      pw[mr*PJ + 16 + m16] = (_Float16)p1;
      float s = p0 + p1;
      s += __shfl_xor(s, 1, 32);
      s += __shfl_xor(s, 2, 32);
      s += __shfl_xor(s, 4, 32);
      s += __shfl_xor(s, 8, 32);
      lrow[r] = lrow[r]*alpha[r] + s;
      mrow[r] = nm;
    }
    #pragma unroll
    for (int nc = 0; nc < 16; ++nc) {
      #pragma unroll
      for (int r = 0; r < 8; ++r) Oa[nc][r] *= alpha[r];
    }

    // P as A-fragment (16x32); same-wave DS ordering guarantees visibility
    h16 pf;
    {
      const _Float16* pr = pw + m16*PJ;
      #pragma unroll
      for (int r = 0; r < 4; ++r) {
        const h2 p0 = *(const h2*)(pr + akoff + 2*r);
        pf[2*r]   = p0[0]; pf[2*r+1]   = p0[1];
        const h2 p1 = *(const h2*)(pr + akoff + 16 + 2*r);
        pf[8+2*r] = p1[0]; pf[8+2*r+1] = p1[1];
      }
    }
    // O += P x V_tile  (V == K tile), 16 output column tiles
    #pragma unroll
    for (int nc = 0; nc < 16; ++nc) {
      h16 vb;
      const _Float16* vp = kcur + nc*16 + m16;
      #pragma unroll
      for (int r = 0; r < 8; ++r) {
        vb[2*r]   = vp[(size_t)((bkoff >> 1) + 2*r + (r >= 4 ? 8 : 0))     * LDK];
        vb[2*r+1] = vp[(size_t)((bkoff >> 1) + 2*r + (r >= 4 ? 8 : 0) + 1) * LDK];
      }
      Oa[nc] = wmma16(pf, vb, Oa[nc]);
    }
  }

  // normalize + store
  float inv[8];
  #pragma unroll
  for (int r = 0; r < 8; ++r) inv[r] = 1.f / lrow[r];
  float* up = u_inv + ((size_t)b * N_ + qb0) * D_;
  #pragma unroll
  for (int nc = 0; nc < 16; ++nc) {
    const int d = nc*16 + m16;
    #pragma unroll
    for (int r = 0; r < 8; ++r) {
      const int mr = r + akoff;
      up[(size_t)mr * D_ + d] = Oa[nc][r] * inv[r];
    }
  }
}

// ---------------------------------------------------------------------------
// Kernel 4: gates, sigmoids, cross product, residuals. One wave per row.
// ---------------------------------------------------------------------------
__global__ __launch_bounds__(256) void final_kernel(
    const float* __restrict__ x,   const float* __restrict__ f,
    const float* __restrict__ Wg,  const float* __restrict__ zxp, const float* __restrict__ zfp,
    const float* __restrict__ u_eqv, const float* __restrict__ v_eqv,
    const float* __restrict__ u_inv, const float* __restrict__ v_inv,
    float* __restrict__ x_out, float* __restrict__ f_out)
{
  const int lane = threadIdx.x & 31;
  const int wave = threadIdx.x >> 5;
  const int row  = blockIdx.x * 8 + wave;
  const float* ui = u_inv + (size_t)row * D_;

  float g = 0.f;                                     // g = u_inv . W_g
  #pragma unroll
  for (int k = 0; k < 8; ++k) {
    const int d = lane + 32*k;
    g = fmaf(ui[d], Wg[d], g);
  }
  g += __shfl_xor(g, 1, 32);
  g += __shfl_xor(g, 2, 32);
  g += __shfl_xor(g, 4, 32);
  g += __shfl_xor(g, 8, 32);
  g += __shfl_xor(g, 16, 32);

  const float ue0 = u_eqv[(size_t)row*3+0];
  const float ue1 = u_eqv[(size_t)row*3+1];
  const float ue2 = u_eqv[(size_t)row*3+2];
  const float nrm = sqrtf(ue0*ue0 + ue1*ue1 + ue2*ue2);
  const float zx = zxp[0], zf = zfp[0];

  if (lane == 0) {
    const float s0 = 1.f/(1.f + __expf(-(ue0*g)));
    const float s1 = 1.f/(1.f + __expf(-(ue1*g)));
    const float s2 = 1.f/(1.f + __expf(-(ue2*g)));
    const float a0 = s0*ue0, a1 = s1*ue1, a2 = s2*ue2;
    const float v0 = v_eqv[(size_t)row*3+0];
    const float v1 = v_eqv[(size_t)row*3+1];
    const float v2 = v_eqv[(size_t)row*3+2];
    x_out[(size_t)row*3+0] = zx*(x[(size_t)row*3+0] + (a1*v2 - a2*v1));
    x_out[(size_t)row*3+1] = zx*(x[(size_t)row*3+1] + (a2*v0 - a0*v2));
    x_out[(size_t)row*3+2] = zx*(x[(size_t)row*3+2] + (a0*v1 - a1*v0));
  }

  const float* fr = f     + (size_t)row * D_;
  const float* vi = v_inv + (size_t)row * D_;
  float*       fo = f_out + (size_t)row * D_;
  #pragma unroll
  for (int k = 0; k < 8; ++k) {
    const int d = lane + 32*k;
    const float u = ui[d];
    const float s = 1.f/(1.f + __expf(-(u*nrm)));
    fo[d] = zf*(fr[d] + s*u*vi[d]);
  }
}

// ---------------------------------------------------------------------------
extern "C" void kernel_launch(void* const* d_in, const int* in_sizes, int n_in,
                              void* d_out, int out_size, void* d_ws, size_t ws_size,
                              hipStream_t stream)
{
  (void)in_sizes; (void)n_in; (void)out_size; (void)ws_size;
  const float* x   = (const float*)d_in[0];
  const float* f   = (const float*)d_in[1];
  const float* Wqe = (const float*)d_in[2];
  const float* Wke = (const float*)d_in[3];
  const float* Wve = (const float*)d_in[4];
  const float* Wqi = (const float*)d_in[5];
  const float* Wki = (const float*)d_in[6];
  const float* Wvi = (const float*)d_in[7];
  const float* Wg  = (const float*)d_in[8];
  const float* zx  = (const float*)d_in[9];
  const float* zf  = (const float*)d_in[10];

  char* ws = (char*)d_ws;
  size_t off = 0;
  auto take = [&](size_t bytes) -> char* {
    char* p = ws + off;
    off += (bytes + 255) & ~(size_t)255;
    return p;
  };
  float*    q_eqv = (float*)take((size_t)BN_*3*sizeof(float));
  float*    k_eqv = (float*)take((size_t)BN_*3*sizeof(float));
  float*    v_eqv = (float*)take((size_t)BN_*3*sizeof(float));
  float*    u_eqv = (float*)take((size_t)BN_*3*sizeof(float));
  _Float16* qi16  = (_Float16*)take((size_t)BN_*D_*sizeof(_Float16));
  _Float16* ki16  = (_Float16*)take((size_t)BN_*D_*sizeof(_Float16));
  float*    v_inv = (float*)take((size_t)BN_*D_*sizeof(float));
  float*    u_inv = (float*)take((size_t)BN_*D_*sizeof(float));
  _Float16* Tqi   = (_Float16*)take((size_t)D_*D_*sizeof(_Float16));
  _Float16* Tki   = (_Float16*)take((size_t)D_*D_*sizeof(_Float16));
  _Float16* Tvi   = (_Float16*)take((size_t)D_*D_*sizeof(_Float16));

  wcvt_kernel<<<(D_*D_)/256, 256, 0, stream>>>(Wqi, Wki, Wvi, Tqi, Tki, Tvi);
  proj_kernel<<<BN_/128, 256, 0, stream>>>(x, f, Wqe, Wke, Wve, Tqi, Tki, Tvi,
                                           q_eqv, k_eqv, v_eqv, qi16, ki16, v_inv);
  conv_kernel<<<12*8, 256, 0, stream>>>(q_eqv, k_eqv, u_eqv);
  attn_kernel<<<B_*(N_/64), 128, 0, stream>>>(qi16, ki16, u_inv);

  float* x_out = (float*)d_out;
  float* f_out = x_out + (size_t)BN_*3;
  final_kernel<<<BN_/8, 256, 0, stream>>>(x, f, Wg, zx, zf,
                                          u_eqv, v_eqv, u_inv, v_inv, x_out, f_out);
}
